// SA_upsample_20598663152418
// MI455X (gfx1250) — matrix operand order repaired and checked by reading, not compiled, not run on hardware
//
#include <hip/hip_runtime.h>
#include <math.h>

// ---------------------------------------------------------------------------
// SA-upsample for MI455X (gfx1250, wave32).
// Hardcoded problem geometry from setup_inputs():
//   B=2, C=64, H=W=192, scale=scale2=2 -> SH=SW=384
// Design: all three big contractions run on v_wmma_f32_16x16x32_f16.
// Memory-bound problem (~300MB @ 23.3TB/s); layouts are pixel-major so every
// WMMA fragment is a contiguous per-lane 16B/32B load and bilinear gathers
// read 256B-contiguous channel rows. Expert mixing uses wave shuffles
// (ds_bpermute) instead of LDS. Periodicity of the coordinate MLP (period 2)
// reduces it to 4 evaluations.
// ---------------------------------------------------------------------------

#define BB   2
#define CC   64
#define HH   192
#define WW   192
#define HWP  (HH*WW)          // 36864 pixels per image
#define SHH  384
#define SWW  384
#define SHW  (SHH*SWW)        // 147456 output pixels per image
#define KC   192              // concat channels
#define G1   (BB*HWP)         // 73728  feat pixels
#define G2   (BB*SHW)         // 294912 output pixels

typedef __attribute__((ext_vector_type(16))) _Float16 v16h;
typedef __attribute__((ext_vector_type(8)))  float    v8f;

union U16H { v16h v; uint4 q[2]; _Float16 h[16]; };
union U8F  { v8f  v; float4 q[2]; float f[8]; };

// ---------------------------------------------------------------------------
// Kernel 1: weight packing (f32 -> f16, WMMA-friendly row-major) + the 4
// periodic variants of the routing/offset MLP.
// ---------------------------------------------------------------------------
__global__ __launch_bounds__(256)
void prep_kernel(const float* __restrict__ wc_e,   // [4,8,64]
                 const float* __restrict__ we_e,   // [4,64,8]
                 const float* __restrict__ wb1, const float* __restrict__ bb1,
                 const float* __restrict__ wb2, const float* __restrict__ bb2,
                 const float* __restrict__ wr,  const float* __restrict__ br,
                 const float* __restrict__ wo,  const float* __restrict__ bo,
                 const float* __restrict__ wconv, // [64,192]
                 _Float16* __restrict__ Wcat,     // [64][192]
                 _Float16* __restrict__ WCall,    // [32][64]  rows = 8e+k
                 _Float16* __restrict__ WEcat,    // [64][32]  cols = 8e+j
                 float* __restrict__ rout4,       // [4][4]
                 float* __restrict__ offs4)       // [4][2]
{
  const int t = threadIdx.x;
  for (int i = t; i < 64*192; i += 256) Wcat[i] = (_Float16)wconv[i];
  // weight_compress [4,8,64] flattens directly to [32][64]
  for (int i = t; i < 32*64; i += 256) WCall[i] = (_Float16)wc_e[i];
  // weight_expand [4,64,8]: WEcat[c][8e+j] = we[e][c][j]
  for (int i = t; i < 4*64*8; i += 256) {
    int e = i >> 9, rem = i & 511, c = rem >> 3, j = rem & 7;
    WEcat[c*32 + e*8 + j] = (_Float16)we_e[i];
  }
  if (t < 4) {
    // variant v = (oy&1)*2 + (ox&1); coor(parity 0) = -0.25, coor(1) = +0.25
    float inp[4] = { 0.5f, 0.5f,
                     (t >> 1) ? 0.25f : -0.25f,
                     (t &  1) ? 0.25f : -0.25f };
    float h1[64], h2[64];
    for (int o = 0; o < 64; ++o) {
      float s = bb1[o];
      for (int i = 0; i < 4; ++i) s += wb1[o*4 + i] * inp[i];
      h1[o] = fmaxf(s, 0.f);
    }
    for (int o = 0; o < 64; ++o) {
      float s = bb2[o];
      for (int i = 0; i < 64; ++i) s += wb2[o*64 + i] * h1[i];
      h2[o] = fmaxf(s, 0.f);
    }
    for (int e = 0; e < 4; ++e) {
      float s = br[e];
      for (int i = 0; i < 64; ++i) s += wr[e*64 + i] * h2[i];
      rout4[t*4 + e] = 1.f / (1.f + expf(-s));
    }
    for (int o = 0; o < 2; ++o) {
      float s = bo[o];
      for (int i = 0; i < 64; ++i) s += wo[o*64 + i] * h2[i];
      offs4[t*2 + o] = s;
    }
  }
}

// ---------------------------------------------------------------------------
// Kernel 2: Sobel + pack concat(x, gx, gy) into pixel-major f16 [g][192].
// ---------------------------------------------------------------------------
__global__ __launch_bounds__(256)
void sobel_pack_kernel(const float* __restrict__ x, _Float16* __restrict__ xcat)
{
  int t  = blockIdx.x * 256 + threadIdx.x;       // (b,c,y,xq), xq fastest
  int xq = t % WW; int r = t / WW;
  int y  = r % HH; r /= HH;
  int c  = r % CC; int b = r / CC;
  const float* xp = x + (size_t)((b*CC + c) * HH) * WW;
  float v[3][3];
#pragma unroll
  for (int dy = 0; dy < 3; ++dy)
#pragma unroll
    for (int dx = 0; dx < 3; ++dx) {
      int yy = y + dy - 1, xx = xq + dx - 1;
      v[dy][dx] = (yy >= 0 && yy < HH && xx >= 0 && xx < WW) ? xp[yy*WW + xx] : 0.f;
    }
  float gx = (v[0][2] - v[0][0]) + 2.f*(v[1][2] - v[1][0]) + (v[2][2] - v[2][0]);
  float gy = (v[2][0] - v[0][0]) + 2.f*(v[2][1] - v[0][1]) + (v[2][2] - v[0][2]);
  _Float16* o = xcat + (size_t)(b*HWP + y*WW + xq) * KC + c;
  o[0]   = (_Float16)v[1][1];
  o[64]  = (_Float16)gx;
  o[128] = (_Float16)gy;
}

// ---------------------------------------------------------------------------
// Kernel 3: feat[g][64] = Wcat[64,192] @ xcat[:,g] + bconv   (WMMA)
// One wave per 16-pixel N-tile; M=64 -> 4 m-tiles; K=192 -> 6 k-steps.
// ---------------------------------------------------------------------------
__global__ __launch_bounds__(256)
void gemm_feat_kernel(const _Float16* __restrict__ Wcat,
                      const _Float16* __restrict__ xcat,
                      const float* __restrict__ bconv,
                      float* __restrict__ feat)
{
  const int lane = threadIdx.x & 31;
  const int tile = blockIdx.x * 8 + (threadIdx.x >> 5);
  const int n0   = tile * 16;
  const int nl   = lane & 15;     // A row / B col within tile
  const int h    = lane >> 4;     // lane half -> K sub-block

  const _Float16* brow = xcat + (size_t)(n0 + nl) * KC + 16 * h;

  U8F acc[4];
#pragma unroll
  for (int mt = 0; mt < 4; ++mt)
#pragma unroll
    for (int r = 0; r < 8; ++r) acc[mt].f[r] = bconv[16*mt + 8*h + r];

#pragma unroll
  for (int ks = 0; ks < 6; ++ks) {
    const int k0 = ks * 32;
    // B fragment: 16 contiguous halves (channels k0+16h .. +15) of this pixel
    U16H bf;
    bf.q[0] = *(const uint4*)(brow + k0);
    bf.q[1] = *(const uint4*)(brow + k0 + 8);
    if (ks < 5) __builtin_prefetch(brow + k0 + 32, 0, 1);   // global_prefetch_b8
#pragma unroll
    for (int mt = 0; mt < 4; ++mt) {
      const _Float16* ar = Wcat + (16*mt + nl) * KC + k0 + 16*h;
      U16H af;
      af.q[0] = *(const uint4*)ar;
      af.q[1] = *(const uint4*)(ar + 8);
      acc[mt].v = __builtin_amdgcn_wmma_f32_16x16x32_f16(
          false, af.v, false, bf.v, (short)0, acc[mt].v, false, false);
    }
  }
  float* orow = feat + (size_t)(n0 + nl) * 64 + 8 * h;
#pragma unroll
  for (int mt = 0; mt < 4; ++mt) {
    *(float4*)(orow + 16*mt)     = acc[mt].q[0];
    *(float4*)(orow + 16*mt + 4) = acc[mt].q[1];
  }
}

// ---------------------------------------------------------------------------
// Kernel 4: bilinear grid-sample (zeros padding) -> fea0[g][64].
// One wave per output pixel; lane handles 2 channels (float2, 256B/row).
// ---------------------------------------------------------------------------
__global__ __launch_bounds__(256)
void bilinear_kernel(const float* __restrict__ feat,
                     const float* __restrict__ offs4,
                     float* __restrict__ fea0)
{
  const int t    = blockIdx.x * 256 + threadIdx.x;
  const int lane = t & 31;
  const int pix  = t >> 5;                 // 0..G2-1
  const int b    = pix / SHW;
  const int q    = pix - b * SHW;
  const int oy   = q / SWW, ox = q - oy * SWW;
  const int var  = ((oy & 1) << 1) | (ox & 1);
  const float offx = offs4[var*2 + 0];
  const float offy = offs4[var*2 + 1];

  // base grid (align_corners=True-style normalization) + learned offsets
  const float sxw = 2.f / (float)(WW - 1);
  const float syh = 2.f / (float)(HH - 1);
  float gx = ((ox + 0.5f) * 0.5f - 0.5f) * sxw - 1.f + offx * sxw;
  float gy = ((oy + 0.5f) * 0.5f - 0.5f) * syh - 1.f + offy * syh;
  float ix = ((gx + 1.f) * WW - 1.f) * 0.5f;
  float iy = ((gy + 1.f) * HH - 1.f) * 0.5f;
  float x0 = floorf(ix), y0 = floorf(iy);
  float wx1 = ix - x0, wx0 = 1.f - wx1;
  float wy1 = iy - y0, wy0 = 1.f - wy1;

  const float* fb = feat + (size_t)b * HWP * 64 + 2 * lane;
  float sx = 0.f, sy = 0.f;
#pragma unroll
  for (int cy = 0; cy < 2; ++cy)
#pragma unroll
    for (int cx = 0; cx < 2; ++cx) {
      float xc = x0 + (float)cx, yc = y0 + (float)cy;
      float wgt = (cx ? wx1 : wx0) * (cy ? wy1 : wy0);
      bool valid = (xc >= 0.f) && (xc <= (float)(WW - 1)) &&
                   (yc >= 0.f) && (yc <= (float)(HH - 1));
      int xi = (int)fminf(fmaxf(xc, 0.f), (float)(WW - 1));
      int yi = (int)fminf(fmaxf(yc, 0.f), (float)(HH - 1));
      float2 v = *(const float2*)(fb + (size_t)(yi*WW + xi) * 64);
      float f = valid ? wgt : 0.f;
      sx += f * v.x; sy += f * v.y;
    }
  float2 s; s.x = sx; s.y = sy;
  *(float2*)(fea0 + (size_t)pix * 64 + 2 * lane) = s;
}

// ---------------------------------------------------------------------------
// Kernel 5: fused MoE compress->mix->expand + residual   (WMMA x2 + shfl)
//   u  = WCall[32,64] @ f          (2 m-tiles, K=64 -> 2 k-steps)
//   m_j = sum_e r_e * u[8e+j]      (cross-half via ds_bpermute shuffles)
//   mm[8e+j] = r_e * m_j
//   out = WEcat[64,32] @ mm + f    (4 m-tiles, K=32, acc pre-loaded with f)
// Output stored NCHW into d_out.
// ---------------------------------------------------------------------------
__global__ __launch_bounds__(256)
void gemm_moe_kernel(const _Float16* __restrict__ WCall,
                     const _Float16* __restrict__ WEcat,
                     const float* __restrict__ fea0,
                     const float* __restrict__ rout4,
                     float* __restrict__ out)
{
  const int lane = threadIdx.x & 31;
  const int tile = blockIdx.x * 8 + (threadIdx.x >> 5);
  const int n0   = tile * 16;
  const int nl   = lane & 15;
  const int h    = lane >> 4;

  const int g  = n0 + nl;
  const int b  = g / SHW;
  const int q  = g - b * SHW;
  const int oy = q / SWW, ox = q - oy * SWW;
  const int var = ((oy & 1) << 1) | (ox & 1);
  const float r0 = rout4[var*4 + 0], r1 = rout4[var*4 + 1];
  const float r2 = rout4[var*4 + 2], r3 = rout4[var*4 + 3];

  const float* frow = fea0 + (size_t)g * 64;

  // ---- GEMM2: u = WCall @ f ----
  U8F u0, u1;
#pragma unroll
  for (int r = 0; r < 8; ++r) { u0.f[r] = 0.f; u1.f[r] = 0.f; }
#pragma unroll
  for (int t2 = 0; t2 < 2; ++t2) {
    const int k0 = t2 * 32;
    const float* bp = frow + k0 + 16 * h;     // 16 contiguous f32 channels
    U16H bf;
#pragma unroll
    for (int e = 0; e < 16; ++e) bf.h[e] = (_Float16)bp[e];
    {
      const _Float16* ar = WCall + (size_t)nl * 64 + k0 + 16*h;
      U16H af; af.q[0] = *(const uint4*)ar; af.q[1] = *(const uint4*)(ar + 8);
      u0.v = __builtin_amdgcn_wmma_f32_16x16x32_f16(
          false, af.v, false, bf.v, (short)0, u0.v, false, false);
    }
    {
      const _Float16* ar = WCall + (size_t)(16 + nl) * 64 + k0 + 16*h;
      U16H af; af.q[0] = *(const uint4*)ar; af.q[1] = *(const uint4*)(ar + 8);
      u1.v = __builtin_amdgcn_wmma_f32_16x16x32_f16(
          false, af.v, false, bf.v, (short)0, u1.v, false, false);
    }
  }

  // ---- expert mix: m_j = r0*u[j] + r1*u[8+j] + r2*u[16+j] + r3*u[24+j] ----
  // row 8e+j lives at lane (nl + 16*(e&1)) of accumulator (e<2 ? u0 : u1)
  float mj[8];
#pragma unroll
  for (int j = 0; j < 8; ++j) {
    float a0 = __shfl(u0.f[j], nl,      32);
    float a1 = __shfl(u0.f[j], nl + 16, 32);
    float a2 = __shfl(u1.f[j], nl,      32);
    float a3 = __shfl(u1.f[j], nl + 16, 32);
    mj[j] = r0*a0 + r1*a1 + r2*a2 + r3*a3;
  }

  // ---- build mm B-fragment: K = 16h + e, mm[K] = r[K>>3] * m[K&7] ----
  U16H bf2;
#pragma unroll
  for (int e = 0; e < 16; ++e) {
    float re = h ? (e < 8 ? r2 : r3) : (e < 8 ? r0 : r1);
    bf2.h[e] = (_Float16)(re * mj[e & 7]);
  }

  // ---- GEMM3 + residual, store NCHW ----
  const size_t obase = (size_t)b * 64 * SHW + q;
#pragma unroll
  for (int mt = 0; mt < 4; ++mt) {
    const _Float16* ar = WEcat + (size_t)(16*mt + nl) * 32 + 16*h;
    U16H af; af.q[0] = *(const uint4*)ar; af.q[1] = *(const uint4*)(ar + 8);
    U8F d;
    const float* fp = frow + 16*mt + 8*h;
    d.q[0] = *(const float4*)fp;
    d.q[1] = *(const float4*)(fp + 4);
    d.v = __builtin_amdgcn_wmma_f32_16x16x32_f16(
        false, af.v, false, bf2.v, (short)0, d.v, false, false);
#pragma unroll
    for (int r = 0; r < 8; ++r)
      out[obase + (size_t)(16*mt + 8*h + r) * SHW] = d.f[r];
  }
}

// ---------------------------------------------------------------------------
// Launch
// ---------------------------------------------------------------------------
extern "C" void kernel_launch(void* const* d_in, const int* in_sizes, int n_in,
                              void* d_out, int out_size, void* d_ws, size_t ws_size,
                              hipStream_t stream) {
  const float* x      = (const float*)d_in[0];
  const float* wc_e   = (const float*)d_in[1];
  const float* we_e   = (const float*)d_in[2];
  const float* wb1    = (const float*)d_in[3];
  const float* bb1    = (const float*)d_in[4];
  const float* wb2    = (const float*)d_in[5];
  const float* bb2    = (const float*)d_in[6];
  const float* wr     = (const float*)d_in[7];
  const float* br     = (const float*)d_in[8];
  const float* wo     = (const float*)d_in[9];
  const float* bo     = (const float*)d_in[10];
  const float* wconv  = (const float*)d_in[11];
  const float* bconv  = (const float*)d_in[12];
  float* outp = (float*)d_out;

  char* ws = (char*)d_ws;
  _Float16* Wcat  = (_Float16*)(ws + 0);                 // 24576 B
  _Float16* WCall = (_Float16*)(ws + 24576);             //  4096 B
  _Float16* WEcat = (_Float16*)(ws + 28672);             //  4096 B
  float*    rout4 = (float*)  (ws + 30720);              //    64 B
  float*    offs4 = (float*)  (ws + 30784);              //    32 B
  _Float16* xcat  = (_Float16*)(ws + 32768);             // 28,311,552 B
  float*    feat  = (float*)  (ws + 32768 + 28311552ull);      // 18,874,368 B
  float*    fea0  = (float*)  (ws + 32768 + 28311552ull + 18874368ull); // 75,497,472 B

  prep_kernel<<<1, 256, 0, stream>>>(wc_e, we_e, wb1, bb1, wb2, bb2,
                                     wr, br, wo, bo, wconv,
                                     Wcat, WCall, WEcat, rout4, offs4);

  sobel_pack_kernel<<<(BB*CC*HH*WW)/256, 256, 0, stream>>>(x, xcat);

  gemm_feat_kernel<<<(G1/16)/8, 256, 0, stream>>>(Wcat, xcat, bconv, feat);

  bilinear_kernel<<<(G2*32)/256, 256, 0, stream>>>(feat, offs4, fea0);

  gemm_moe_kernel<<<(G2/16)/8, 256, 0, stream>>>(WCall, WEcat, fea0, rout4, outp);
}
